// GAT_16157666968018
// MI455X (gfx1250) — matrix operand order repaired and checked by reading, not compiled, not run on hardware
//
#include <hip/hip_runtime.h>
#include <hip/hip_bf16.h>
#include <math.h>

#define NNODES 20000
#define NE     640000
#define NETOT  (NE + NNODES)   // edges + self loops
#define NH     4
#define NC     64
#define NF     256             // H*C
#define F0     128
#define NG     64
#define NEGS   0.2f

typedef __attribute__((ext_vector_type(16))) __bf16 v16bf;
typedef __attribute__((ext_vector_type(8)))  __bf16 v8bf;
typedef __attribute__((ext_vector_type(8)))  float  v8f;

__device__ __forceinline__ unsigned short f2bf(float f) {
  unsigned u = __float_as_uint(f);
  u += 0x7fffu + ((u >> 16) & 1u);       // round-to-nearest-even
  return (unsigned short)(u >> 16);
}
// monotone uint encoding of float for atomicMax
__device__ __forceinline__ unsigned fenc(float f) {
  unsigned u = __float_as_uint(f);
  return (u & 0x80000000u) ? ~u : (u | 0x80000000u);
}
__device__ __forceinline__ float fdec(unsigned u) {
  u = (u & 0x80000000u) ? (u & 0x7fffffffu) : ~u;
  return __uint_as_float(u);
}

__global__ void k_zero(float* p, int n) {
  int i = blockIdx.x * blockDim.x + threadIdx.x;
  if (i < n) p[i] = 0.0f;
}
__global__ void k_fill_u32(unsigned* p, unsigned v, int n) {
  int i = blockIdx.x * blockDim.x + threadIdx.x;
  if (i < n) p[i] = v;
}
__global__ void k_f32_to_bf16(const float* __restrict__ s, unsigned short* __restrict__ d, int n) {
  int i = blockIdx.x * blockDim.x + threadIdx.x;
  if (i < n) d[i] = f2bf(s[i]);
}
// W[K,NF] f32 -> Wt[NF,K] bf16
__global__ void k_transpose_w(const float* __restrict__ W, unsigned short* __restrict__ Wt, int K) {
  int i = blockIdx.x * blockDim.x + threadIdx.x;
  if (i >= K * NF) return;
  int k = i / NF, o = i % NF;
  Wt[o * K + k] = f2bf(W[i]);
}

// h[M,NF] = A[M,K](bf16) x Wt[NF,K](bf16, pre-transposed), one 16x16 tile per wave
template <int K>
__global__ __launch_bounds__(256) void k_gemm_bf16(const unsigned short* __restrict__ A,
                                                   const unsigned short* __restrict__ Bt,
                                                   float* __restrict__ D, int M) {
  int wave  = (blockIdx.x << 3) + (threadIdx.x >> 5);
  int lane  = threadIdx.x & 31;
  int tileN = wave & 15;                 // NF/16 = 16 tiles across
  int tileM = wave >> 4;
  if (tileM * 16 >= M) return;

  // A fragment (16x32 bf16): lanes<16 hold K=k..k+7 & k+16..k+23 of row M=lane;
  // lanes>=16 hold the +8 shifted chunks (ISA 7.12.2 layout).
  int rowA = tileM * 16 + (lane & 15);
  int colB = tileN * 16 + (lane & 15);
  int hiA  = (lane >> 4) * 8;
  int hiB  = (lane >> 4) * 8;            // B(32x16): same K chunks as A (lanes>=16 shifted by +8)
  const unsigned short* pa = A  + (size_t)rowA * K + hiA;
  const unsigned short* pb = Bt + (size_t)colB * K + hiB;

  v8f acc = {};
#pragma unroll
  for (int k = 0; k < K; k += 32) {
    v8bf a0 = *(const v8bf*)(pa + k);
    v8bf a1 = *(const v8bf*)(pa + k + 16);
    v8bf b0 = *(const v8bf*)(pb + k);
    v8bf b1 = *(const v8bf*)(pb + k + 16);
    v16bf av = __builtin_shufflevector(a0, a1, 0,1,2,3,4,5,6,7,8,9,10,11,12,13,14,15);
    v16bf bv = __builtin_shufflevector(b0, b1, 0,1,2,3,4,5,6,7,8,9,10,11,12,13,14,15);
    acc = __builtin_amdgcn_wmma_f32_16x16x32_bf16(false, av, false, bv,
                                                  (short)0, acc, false, false);
  }
  // D layout: VGPR r -> M = r (lanes 0-15) / r+8 (lanes 16-31), N = lane&15
  int row0 = tileM * 16 + (lane >> 4) * 8;
  int col  = tileN * 16 + (lane & 15);
  float* out = D + (size_t)row0 * NF + col;
#pragma unroll
  for (int r = 0; r < 8; ++r) out[(size_t)r * NF] = acc[r];
}

// per-(node,head) attention logits: as_=sum_c h*a_s, ad_=sum_c h*a_d
__global__ void k_alpha(const float* __restrict__ h, const float* __restrict__ a_s,
                        const float* __restrict__ a_d, float* __restrict__ as_,
                        float* __restrict__ ad_) {
  int i = blockIdx.x * blockDim.x + threadIdx.x;
  if (i >= NNODES * NH) return;
  int hh = i & 3;
  const float* hp = h + (size_t)(i >> 2) * NF + hh * NC;
  float ss = 0.f, sd = 0.f;
#pragma unroll 8
  for (int c = 0; c < NC; ++c) {
    float v = hp[c];
    ss += v * a_s[hh * NC + c];
    sd += v * a_d[hh * NC + c];
  }
  as_[i] = ss;
  ad_[i] = sd;
}

// segment max over incoming edges (bit-encoded float atomicMax)
__global__ void k_edge_max(const int* __restrict__ src, const int* __restrict__ dst,
                           const float* __restrict__ as_, const float* __restrict__ ad_,
                           unsigned* __restrict__ m) {
  int i = blockIdx.x * blockDim.x + threadIdx.x;
  if (i >= NETOT * NH) return;
  int e = i >> 2, hh = i & 3;
  int s, t;
  if (e < NE) { s = src[e]; t = dst[e]; } else { s = e - NE; t = s; }
  float x = as_[s * NH + hh] + ad_[t * NH + hh];
  x = x > 0.f ? x : NEGS * x;
  atomicMax(&m[t * NH + hh], fenc(x));
}

// one wave per edge: den[dst,h] += al; agg[dst,h,c] += al * h[src,h,c]
__global__ __launch_bounds__(256) void k_edge_acc(const int* __restrict__ src,
                                                  const int* __restrict__ dst,
                                                  const float* __restrict__ as_,
                                                  const float* __restrict__ ad_,
                                                  const unsigned* __restrict__ m,
                                                  const float* __restrict__ h,
                                                  float* __restrict__ den,
                                                  float* __restrict__ agg) {
  int wave = (int)(((unsigned)blockIdx.x * blockDim.x + threadIdx.x) >> 5);
  int lane = threadIdx.x & 31;
  if (wave >= NETOT) return;
  int s, t;
  if (wave < NE) { s = src[wave]; t = dst[wave]; } else { s = wave - NE; t = s; }
  float al[NH];
#pragma unroll
  for (int hh = 0; hh < NH; ++hh) {
    float x = as_[s * NH + hh] + ad_[t * NH + hh];
    x = x > 0.f ? x : NEGS * x;
    al[hh] = __expf(x - fdec(m[t * NH + hh]));
  }
#pragma unroll
  for (int hh = 0; hh < NH; ++hh)
    if (lane == hh) atomicAdd(&den[t * NH + hh], al[hh]);
  const float* hp = h + (size_t)s * NF;
  float* ap = agg + (size_t)t * NF;
#pragma unroll
  for (int j = 0; j < 8; ++j) {           // head index j>>1 is compile-time constant
    int idx = j * 32 + lane;
    atomicAdd(&ap[idx], al[j >> 1] * hp[idx]);
  }
}

// x_next = elu(agg/den + b); emit fp32 (for gather/pool) and bf16 (next GEMM A)
__global__ void k_finalize(const float* __restrict__ agg, const float* __restrict__ den,
                           const float* __restrict__ b, float* __restrict__ xf,
                           unsigned short* __restrict__ xb) {
  int i = blockIdx.x * blockDim.x + threadIdx.x;
  if (i >= NNODES * NF) return;
  int idx = i & (NF - 1);
  int n   = i >> 8;
  float v = agg[i] / den[n * NH + (idx >> 6)] + b[idx];
  v = v > 0.f ? v : expm1f(v);
  xf[i] = v;
  xb[i] = f2bf(v);
}

__global__ void k_pool(const float* __restrict__ xf, const int* __restrict__ batch,
                       float* __restrict__ pool) {
  int i = blockIdx.x * blockDim.x + threadIdx.x;
  if (i >= NNODES * NF) return;
  int n = i >> 8;
  atomicAdd(&pool[batch[n] * NF + (i & 255)], xf[i]);
}
__global__ void k_cnt(const int* __restrict__ batch, float* __restrict__ cnt) {
  int n = blockIdx.x * blockDim.x + threadIdx.x;
  if (n < NNODES) atomicAdd(&cnt[batch[n]], 1.0f);
}

__global__ __launch_bounds__(64) void k_mlp(const float* __restrict__ pool,
                                            const float* __restrict__ cnt,
                                            const float* __restrict__ Wm1,
                                            const float* __restrict__ bm1,
                                            const float* __restrict__ Wm2,
                                            const float* __restrict__ bm2,
                                            float* __restrict__ out) {
  __shared__ float gv[NF];
  __shared__ float hd[64];
  int g = blockIdx.x, t = threadIdx.x;
  float c = fmaxf(cnt[g], 1.0f);
  for (int i = t; i < NF; i += 64) gv[i] = pool[g * NF + i] / c;
  __syncthreads();
  float acc = bm1[t];
  for (int k = 0; k < NF; ++k) acc += gv[k] * Wm1[k * 64 + t];
  hd[t] = fmaxf(acc, 0.f);
  __syncthreads();
  if (t < 2) {
    float o = bm2[t];
    for (int j = 0; j < 64; ++j) o += hd[j] * Wm2[j * 2 + t];
    out[g * 2 + t] = o;
  }
}

extern "C" void kernel_launch(void* const* d_in, const int* in_sizes, int n_in,
                              void* d_out, int out_size, void* d_ws, size_t ws_size,
                              hipStream_t stream) {
  (void)in_sizes; (void)n_in; (void)out_size; (void)ws_size;
  const float* x     = (const float*)d_in[0];
  const int*   ei    = (const int*)d_in[1];
  const int*   batch = (const int*)d_in[2];
  const float* W[3]  = {(const float*)d_in[3], (const float*)d_in[7],  (const float*)d_in[11]};
  const float* As[3] = {(const float*)d_in[4], (const float*)d_in[8],  (const float*)d_in[12]};
  const float* Ad[3] = {(const float*)d_in[5], (const float*)d_in[9],  (const float*)d_in[13]};
  const float* Bi[3] = {(const float*)d_in[6], (const float*)d_in[10], (const float*)d_in[14]};
  const float* Wm1 = (const float*)d_in[15];
  const float* bm1 = (const float*)d_in[16];
  const float* Wm2 = (const float*)d_in[17];
  const float* bm2 = (const float*)d_in[18];
  const int* srcI = ei;
  const int* dstI = ei + NE;

  char* p = (char*)d_ws;
  auto carve = [&](size_t bytes) -> void* {
    void* r = (void*)p;
    p += (bytes + 255) & ~(size_t)255;
    return r;
  };
  unsigned short* xb  = (unsigned short*)carve((size_t)NNODES * NF * 2);
  unsigned short* Wt  = (unsigned short*)carve((size_t)NF * NF * 2);
  float*    hbuf = (float*)carve((size_t)NNODES * NF * 4);
  float*    as_  = (float*)carve((size_t)NNODES * NH * 4);
  float*    ad_  = (float*)carve((size_t)NNODES * NH * 4);
  unsigned* m    = (unsigned*)carve((size_t)NNODES * NH * 4);
  float*    den  = (float*)carve((size_t)NNODES * NH * 4);
  float*    agg  = (float*)carve((size_t)NNODES * NF * 4);
  float*    xf   = (float*)carve((size_t)NNODES * NF * 4);
  float*    pool = (float*)carve((size_t)NG * NF * 4);
  float*    cnt  = (float*)carve((size_t)NG * 4);

  const int T = 256;
  // layer-0 input to bf16
  k_f32_to_bf16<<<(NNODES * F0 + T - 1) / T, T, 0, stream>>>(x, xb, NNODES * F0);

  const int gemmBlocks = (NNODES / 16) * (NF / 16) / 8;   // 1 wave per 16x16 tile, 8 waves/block
  for (int L = 0; L < 3; ++L) {
    const int K = (L == 0) ? F0 : NF;
    k_transpose_w<<<(K * NF + T - 1) / T, T, 0, stream>>>(W[L], Wt, K);
    if (K == F0)
      k_gemm_bf16<F0><<<gemmBlocks, T, 0, stream>>>(xb, Wt, hbuf, NNODES);
    else
      k_gemm_bf16<NF><<<gemmBlocks, T, 0, stream>>>(xb, Wt, hbuf, NNODES);
    k_alpha<<<(NNODES * NH + T - 1) / T, T, 0, stream>>>(hbuf, As[L], Ad[L], as_, ad_);
    k_fill_u32<<<(NNODES * NH + T - 1) / T, T, 0, stream>>>(m, 0x007FFFFFu, NNODES * NH); // enc(-inf)
    k_zero<<<(NNODES * NH + T - 1) / T, T, 0, stream>>>(den, NNODES * NH);
    k_zero<<<(NNODES * NF + T - 1) / T, T, 0, stream>>>(agg, NNODES * NF);
    k_edge_max<<<((size_t)NETOT * NH + T - 1) / T, T, 0, stream>>>(srcI, dstI, as_, ad_, m);
    k_edge_acc<<<((size_t)NETOT * 32 + T - 1) / T, T, 0, stream>>>(srcI, dstI, as_, ad_, m,
                                                                   hbuf, den, agg);
    k_finalize<<<(NNODES * NF + T - 1) / T, T, 0, stream>>>(agg, den, Bi[L], xf, xb);
  }

  k_zero<<<(NG * NF + T - 1) / T, T, 0, stream>>>(pool, NG * NF);
  k_zero<<<(NG + T - 1) / T, T, 0, stream>>>(cnt, NG);
  k_pool<<<(NNODES * NF + T - 1) / T, T, 0, stream>>>(xf, batch, pool);
  k_cnt<<<(NNODES + T - 1) / T, T, 0, stream>>>(batch, cnt);
  k_mlp<<<NG, 64, 0, stream>>>(pool, cnt, Wm1, bm1, Wm2, bm2, (float*)d_out);
}
